// CausalSelfAttention_24704651887043
// MI455X (gfx1250) — matrix operand order repaired and checked by reading, not compiled, not run on hardware
//
#include <hip/hip_runtime.h>
#include <math.h>

typedef __attribute__((ext_vector_type(16))) __bf16        bf16x16;
typedef __attribute__((ext_vector_type(8)))  float         f32x8;
typedef __attribute__((ext_vector_type(8)))  unsigned int  u32x8;

#define D_MODEL   1024
#define N_HEADS   16
#define HEAD_DIM  64
#define SEQ       2048
#define BATCH     2
#define ROWS      (BATCH * SEQ)      /* 4096 */
#define QKV_N     (3 * D_MODEL)      /* 3072 */

static __device__ __forceinline__ unsigned short f2bfbits(float f) {
  unsigned x = __builtin_bit_cast(unsigned, f);
  unsigned r = (x + 0x7FFFu + ((x >> 16) & 1u)) >> 16;  // round-to-nearest-even
  return (unsigned short)r;
}

static __device__ __forceinline__ f32x8 wmma_bf16(bf16x16 a, bf16x16 b, f32x8 c) {
  return __builtin_amdgcn_wmma_f32_16x16x32_bf16(false, a, false, b, (short)0, c,
                                                 false, false);
}

// Async DMA: copy 16 bytes/lane from global to LDS (ASYNCcnt-tracked).
static __device__ __forceinline__ void async_b128(unsigned lds_off,
                                                  const unsigned short* g) {
  asm volatile("global_load_async_to_lds_b128 %0, %1, off"
               :: "v"(lds_off), "v"(g) : "memory");
}
static __device__ __forceinline__ void wait_async_le4() {
  asm volatile("s_wait_asynccnt 0x4" ::: "memory");
}
static __device__ __forceinline__ void wait_async_0() {
  asm volatile("s_wait_asynccnt 0x0" ::: "memory");
}
static __device__ __forceinline__ unsigned lds_off32(const void* p) {
  return (unsigned)(uintptr_t)p;   // LDS aperture: addr[31:0] is the LDS offset
}

// A-frag (16x32, row-major src with K contiguous) or B-frag from K-transposed
// weights ([N,K] row-major): lane m=lane%16 row, lane/16 selects +8 K offset.
// Pairs map to two contiguous 16B chunks -> b128 loads (global or ds).
static __device__ __forceinline__ bf16x16 load_frag_k32(const unsigned short* p, int ld) {
  int lane = threadIdx.x & 31;
  int m = lane & 15, half = lane >> 4;
  const unsigned short* row = p + (size_t)m * ld + half * 8;
  u32x8 r;
#pragma unroll
  for (int v = 0; v < 8; ++v) {
    int k = (v >> 2) * 16 + (v & 3) * 2;
    r[v] = *(const unsigned int*)(row + k);
  }
  return __builtin_bit_cast(bf16x16, r);
}

// B-frag from row-major [K, ld] (e.g. V: keys x headdim) — K pairs strided.
static __device__ __forceinline__ bf16x16 load_fragB_strided(const unsigned short* p, int ld) {
  int lane = threadIdx.x & 31;
  int n = lane & 15, half = lane >> 4;
  u32x8 r;
#pragma unroll
  for (int v = 0; v < 8; ++v) {
    int k = (v >> 2) * 16 + half * 8 + (v & 3) * 2;
    unsigned lo = p[(size_t)k * ld + n];
    unsigned hi = p[(size_t)(k + 1) * ld + n];
    r[v] = lo | (hi << 16);
  }
  return __builtin_bit_cast(bf16x16, r);
}

static __device__ __forceinline__ float rowmax16(float v) {
#pragma unroll
  for (int m = 1; m < 16; m <<= 1) v = fmaxf(v, __shfl_xor(v, m, 32));
  return v;
}
static __device__ __forceinline__ float rowsum16(float v) {
#pragma unroll
  for (int m = 1; m < 16; m <<= 1) v += __shfl_xor(v, m, 32);
  return v;
}

// ---------------- conversion kernels ----------------
__global__ void cvt_f32_bf16(const float* __restrict__ in, unsigned short* __restrict__ out, int n) {
  int i = blockIdx.x * blockDim.x + threadIdx.x;
  if (i < n) out[i] = f2bfbits(in[i]);
}

// in: [K,N] f32 row-major  ->  out: [N,K] bf16 row-major
__global__ void transpose_cvt(const float* __restrict__ in, unsigned short* __restrict__ out,
                              int K, int N) {
  int i = blockIdx.x * blockDim.x + threadIdx.x;
  if (i < K * N) {
    int k = i / N, n = i % N;
    out[(size_t)n * K + k] = f2bfbits(in[i]);
  }
}

// -------- WMMA GEMM with async-LDS double buffering --------
// C[M,N] = A[M,K] @ Bt[N,K]^T + bias
// block = 128 threads (4 waves); block tile 64x64; wave tile 32x32; K step 32.
// Per K step each thread issues 4 global_load_async_to_lds_b128 (2 A + 2 B);
// the wait s_wait_asynccnt<=4 only covers the tile being consumed, so tile k+1
// streams in while tile k feeds the WMMA pipe.
template <bool BF16OUT>
__global__ void gemm_wmma(const unsigned short* __restrict__ A,
                          const unsigned short* __restrict__ Bt,
                          const float* __restrict__ bias,
                          void* __restrict__ Cout, int M, int N, int K) {
  __shared__ unsigned short Atile[2][64 * 32];  // 8 KiB
  __shared__ unsigned short Btile[2][64 * 32];  // 8 KiB

  int t    = threadIdx.x;
  int wid  = t >> 5;
  int lane = t & 31;
  int n_l = lane & 15, half = lane >> 4;

  int tilesM = M >> 6;
  int bm = blockIdx.x % tilesM;
  int bn = blockIdx.x / tilesM;
  int blkM = bm * 64, blkN = bn * 64;

  // this thread's DMA chunk: chunks 4*row+(t&1) and 4*row+(t&1)+2 of 256 (row = 16B-chunk/4)
  int c0  = t * 2;
  int row = c0 >> 2;        // 0..63
  int seg = (c0 & 2) * 4;   // element offset of first 16B chunk (0 or 8)

  const unsigned short* gA = A  + (size_t)(blkM + row) * K + seg;
  const unsigned short* gB = Bt + (size_t)(blkN + row) * K + seg;
  unsigned ldsA = lds_off32(&Atile[0][row * 32 + seg]);
  unsigned ldsB = lds_off32(&Btile[0][row * 32 + seg]);
  const unsigned bufStride = 64 * 32 * 2;  // bytes between double buffers

  f32x8 acc[2][2];
#pragma unroll
  for (int i = 0; i < 2; ++i)
#pragma unroll
    for (int j = 0; j < 2; ++j) acc[i][j] = (f32x8)0.f;

  const int KT = K >> 5;

  // prologue: fill buffer 0 with K-tile 0
  async_b128(ldsA, gA);
  async_b128(ldsA + 32, gA + 16);
  async_b128(ldsB, gB);
  async_b128(ldsB + 32, gB + 16);

  for (int kt = 0; kt < KT; ++kt) {
    int cur = kt & 1;
    if (kt + 1 < KT) {               // prefetch next K-tile into other buffer
      int nb = cur ^ 1;
      const unsigned short* nA = gA + (size_t)(kt + 1) * 32;
      const unsigned short* nB = gB + (size_t)(kt + 1) * 32;
      async_b128(ldsA + nb * bufStride, nA);
      async_b128(ldsA + nb * bufStride + 32, nA + 16);
      async_b128(ldsB + nb * bufStride, nB);
      async_b128(ldsB + nb * bufStride + 32, nB + 16);
      wait_async_le4();              // current tile's 4 ops complete
    } else {
      wait_async_0();
    }
    __syncthreads();                 // all waves' DMA landed

    const unsigned short* As = &Atile[cur][((wid >> 1) * 32) * 32];
    const unsigned short* Bs = &Btile[cur][((wid & 1) * 32) * 32];
    bf16x16 a0 = load_frag_k32(As, 32);
    bf16x16 a1 = load_frag_k32(As + 16 * 32, 32);
    bf16x16 b0 = load_frag_k32(Bs, 32);
    bf16x16 b1 = load_frag_k32(Bs + 16 * 32, 32);
    acc[0][0] = wmma_bf16(a0, b0, acc[0][0]);
    acc[0][1] = wmma_bf16(a0, b1, acc[0][1]);
    acc[1][0] = wmma_bf16(a1, b0, acc[1][0]);
    acc[1][1] = wmma_bf16(a1, b1, acc[1][1]);

    __syncthreads();                 // don't let DMA overwrite a tile in use
  }

  int m0 = blkM + (wid >> 1) * 32;
  int n0 = blkN + (wid & 1) * 32;
#pragma unroll
  for (int i = 0; i < 2; ++i)
#pragma unroll
    for (int j = 0; j < 2; ++j) {
      int col = n0 + j * 16 + n_l;
      float bv = bias[col];
#pragma unroll
      for (int r = 0; r < 8; ++r) {
        int rowo = m0 + i * 16 + r + 8 * half;
        float v = acc[i][j][r] + bv;
        if (BF16OUT)
          ((unsigned short*)Cout)[(size_t)rowo * N + col] = f2bfbits(v);
        else
          ((float*)Cout)[(size_t)rowo * N + col] = v;
      }
    }
}

// ---------------- flash attention: one wave per 16-row Q block ----------------
// qkv: [4096, 3072] bf16 (Q | K | V per row). out: [4096, 1024] bf16.
__global__ void attn_wmma(const unsigned short* __restrict__ qkv,
                          unsigned short* __restrict__ aout) {
  __shared__ unsigned short Ptile[4][16 * 32];  // per-wave P staging (C-layout -> A-layout)

  int wid = threadIdx.x >> 5;
  int lane = threadIdx.x & 31;
  int n_l = lane & 15, half = lane >> 4;

  int gid = blockIdx.x * 4 + wid;               // 0 .. 4095
  int qb = gid & 127;
  int h  = (gid >> 7) & 15;
  int b  = gid >> 11;
  int q0 = qb * 16;

  const unsigned short* Qp = qkv + (size_t)(b * SEQ) * QKV_N + h * HEAD_DIM;
  const unsigned short* Kp = Qp + D_MODEL;
  const unsigned short* Vp = Qp + 2 * D_MODEL;

  bf16x16 aq0 = load_frag_k32(Qp + (size_t)q0 * QKV_N, QKV_N);       // dims 0..31
  bf16x16 aq1 = load_frag_k32(Qp + (size_t)q0 * QKV_N + 32, QKV_N);  // dims 32..63

  f32x8 o[4];
#pragma unroll
  for (int c = 0; c < 4; ++c) o[c] = (f32x8)0.f;
  float m_run[8], l_run[8];
#pragma unroll
  for (int r = 0; r < 8; ++r) { m_run[r] = -INFINITY; l_run[r] = 0.f; }

  const float scale = 0.125f;  // 1/sqrt(64)

  for (int j0 = 0; j0 < q0 + 16; j0 += 32) {
    // ---- scores: S = Q @ K^T  (two 16x16 tiles, keys j0..j0+31) ----
    f32x8 s0 = (f32x8)0.f, s1 = (f32x8)0.f;
    s0 = wmma_bf16(aq0, load_frag_k32(Kp + (size_t)j0 * QKV_N, QKV_N), s0);
    s0 = wmma_bf16(aq1, load_frag_k32(Kp + (size_t)j0 * QKV_N + 32, QKV_N), s0);
    s1 = wmma_bf16(aq0, load_frag_k32(Kp + (size_t)(j0 + 16) * QKV_N, QKV_N), s1);
    s1 = wmma_bf16(aq1, load_frag_k32(Kp + (size_t)(j0 + 16) * QKV_N + 32, QKV_N), s1);

    unsigned short* Pt = &Ptile[wid][0];
#pragma unroll
    for (int r = 0; r < 8; ++r) {
      int qrow = q0 + r + 8 * half;
      int kcol = j0 + n_l;
      float v0 = (kcol <= qrow) ? s0[r] * scale : -INFINITY;
      float v1 = (kcol + 16 <= qrow) ? s1[r] * scale : -INFINITY;

      float mt   = rowmax16(fmaxf(v0, v1));
      float mnew = fmaxf(m_run[r], mt);
      float alpha = __expf(m_run[r] - mnew);
      float p0 = __expf(v0 - mnew);
      float p1 = __expf(v1 - mnew);
      l_run[r] = l_run[r] * alpha + rowsum16(p0 + p1);
      m_run[r] = mnew;

      int row = r + 8 * half;
      Pt[row * 32 + n_l]      = f2bfbits(p0);
      Pt[row * 32 + n_l + 16] = f2bfbits(p1);

      o[0][r] *= alpha; o[1][r] *= alpha; o[2][r] *= alpha; o[3][r] *= alpha;
    }
    asm volatile("" ::: "memory");  // order LDS store -> LDS load (same wave, DS in-order)

    // ---- reload P as A-fragment (16x32) from LDS ----
    u32x8 pr;
    const unsigned short* prow = &Ptile[wid][(lane & 15) * 32];
#pragma unroll
    for (int v = 0; v < 8; ++v) {
      int k = (v >> 2) * 16 + half * 8 + (v & 3) * 2;
      pr[v] = *(const unsigned int*)(prow + k);
    }
    bf16x16 aP = __builtin_bit_cast(bf16x16, pr);

    // ---- O += P @ V  (V block: 32 keys x 64 dims) ----
#pragma unroll
    for (int c = 0; c < 4; ++c) {
      bf16x16 bV = load_fragB_strided(Vp + (size_t)j0 * QKV_N + c * 16, QKV_N);
      o[c] = wmma_bf16(aP, bV, o[c]);
    }
  }

  // ---- epilogue: normalize, store bf16 [4096, 1024] ----
#pragma unroll
  for (int r = 0; r < 8; ++r) {
    float inv = 1.0f / l_run[r];
    int row = b * SEQ + q0 + r + 8 * half;
#pragma unroll
    for (int c = 0; c < 4; ++c) {
      int col = h * HEAD_DIM + c * 16 + n_l;
      aout[(size_t)row * D_MODEL + col] = f2bfbits(o[c][r] * inv);
    }
  }
}

// ---------------- host launch ----------------
extern "C" void kernel_launch(void* const* d_in, const int* in_sizes, int n_in,
                              void* d_out, int out_size, void* d_ws, size_t ws_size,
                              hipStream_t stream) {
  const float* x      = (const float*)d_in[0];  // [2,2048,1024]
  const float* w_qkv  = (const float*)d_in[1];  // [1024,3072]
  const float* b_qkv  = (const float*)d_in[2];  // [3072]
  const float* w_proj = (const float*)d_in[3];  // [1024,1024]
  const float* b_proj = (const float*)d_in[4];  // [1024]
  float* out = (float*)d_out;                   // [2,2048,1024]

  char* ws = (char*)d_ws;
  unsigned short* Xb     = (unsigned short*)(ws);                         // 8 MiB
  unsigned short* WqkvT  = (unsigned short*)(ws + (8u << 20));            // 6 MiB
  unsigned short* WprojT = (unsigned short*)(ws + (14u << 20));           // 2 MiB
  unsigned short* QKV    = (unsigned short*)(ws + (16u << 20));           // 24 MiB
  unsigned short* Attn   = (unsigned short*)(ws + (40u << 20));           // 8 MiB

  {
    int n = ROWS * D_MODEL;
    cvt_f32_bf16<<<(n + 255) / 256, 256, 0, stream>>>(x, Xb, n);
  }
  {
    int n = D_MODEL * QKV_N;
    transpose_cvt<<<(n + 255) / 256, 256, 0, stream>>>(w_qkv, WqkvT, D_MODEL, QKV_N);
    n = D_MODEL * D_MODEL;
    transpose_cvt<<<(n + 255) / 256, 256, 0, stream>>>(w_proj, WprojT, D_MODEL, D_MODEL);
  }
  {
    dim3 grid((ROWS / 64) * (QKV_N / 64));  // 3072 blocks
    gemm_wmma<true><<<grid, 128, 0, stream>>>(Xb, WqkvT, b_qkv, QKV, ROWS, QKV_N, D_MODEL);
  }
  {
    attn_wmma<<<1024, 128, 0, stream>>>(QKV, Attn);
  }
  {
    dim3 grid((ROWS / 64) * (D_MODEL / 64));  // 1024 blocks
    gemm_wmma<false><<<grid, 128, 0, stream>>>(Attn, WprojT, b_proj, out, ROWS, D_MODEL, D_MODEL);
  }
}
